// simple_EF_block_34428457845433
// MI455X (gfx1250) — matrix operand order repaired and checked by reading, not compiled, not run on hardware
//
#include <hip/hip_runtime.h>

// ---- CDNA5 bf16x3 WMMA implicit-GEMM 3x3 conv (fp32 in/out, ~fp32 acc) ----
// Per wave: D[16co x 16w] += A[16co x 32ci] * B[32ci x 16w] via
// V_WMMA_F32_16X16X32_BF16, with A,B split into bf16 hi/lo planes:
//   A*B ~= Ahi*Bhi + Ahi*Blo + Alo*Bhi   (f32 accumulate)
// Block tile: 128 co x 64 w for one (b,h); weights staged via async
// global->LDS (ASYNCcnt, global_load_async_to_lds_b64).

typedef __attribute__((ext_vector_type(4)))  __bf16 v4bf;
typedef __attribute__((ext_vector_type(8)))  __bf16 v8bf;
typedef __attribute__((ext_vector_type(16))) __bf16 v16bf;
typedef __attribute__((ext_vector_type(8)))  float  v8f;
typedef int v2i __attribute__((vector_size(8)));   // type async builtin expects

#define C_CH   128
#define HW     128
#define BATCH  16
#define PLANE  (HW * HW)          // 16384
#define TILE_W 64                 // output w per block (4 N-tiles of 16)
#define XROWS  (TILE_W + 2)       // 66 halo'd input columns
#define CIP    36                 // LDS ci pitch: 8B-aligned b64 frags,
                                  // 72B lane stride -> conflict-free banks
#define WROWS  (3 * C_CH)         // (s, co) rows staged per pass = 384
#define NWEL   (C_CH * C_CH * 9)  // 147456 weight elements

#define AS1 __attribute__((address_space(1)))
#define AS3 __attribute__((address_space(3)))
typedef AS1 v2i* gasync_ptr;
typedef AS3 v2i* lasync_ptr;

#if defined(__HIP_DEVICE_COMPILE__) && __has_builtin(__builtin_amdgcn_global_load_async_to_lds_b64)
#define USE_ASYNC_LDS 1
#else
#define USE_ASYNC_LDS 0
#endif

// Repack weights w[co][ci][r,s] -> bf16 hi/lo planes, LDS-ready layout:
//   dst[(ci/32)][r][s][co][ci%32]
__global__ void repack_w_bf16_kernel(const float* __restrict__ w,
                                     __bf16* __restrict__ wph,
                                     __bf16* __restrict__ wpl) {
    int idx = blockIdx.x * blockDim.x + threadIdx.x;   // co*1152 + ci*9 + tap
    if (idx < NWEL) {
        int co  = idx / (C_CH * 9);
        int rem = idx % (C_CH * 9);
        int ci  = rem / 9;
        int tap = rem % 9;
        int r = tap / 3, s = tap % 3;
        float v = w[idx];
        __bf16 hi = (__bf16)v;
        __bf16 lo = (__bf16)(v - (float)hi);
        int dst = ((((ci >> 5) * 3 + r) * 3 + s) * C_CH + co) * 32 + (ci & 31);
        wph[dst] = hi;
        wpl[dst] = lo;
    }
}

// Build v16bf fragment from two 8-element contiguous runs (4x ds_load_b64).
__device__ __forceinline__ v16bf ld_frag16(const __bf16* p0, const __bf16* p1) {
    v4bf a0 = *(const v4bf*)(p0);
    v4bf a1 = *(const v4bf*)(p0 + 4);
    v4bf a2 = *(const v4bf*)(p1);
    v4bf a3 = *(const v4bf*)(p1 + 4);
    v8bf h0 = __builtin_shufflevector(a0, a1, 0, 1, 2, 3, 4, 5, 6, 7);
    v8bf h1 = __builtin_shufflevector(a2, a3, 0, 1, 2, 3, 4, 5, 6, 7);
    return __builtin_shufflevector(h0, h1, 0, 1, 2, 3, 4, 5, 6, 7,
                                   8, 9, 10, 11, 12, 13, 14, 15);
}

// MODE 0: out = relu(conv + bias)
// MODE 1: out = resid + (1/3)*(conv + bias)
// MODE 2: out = relu(resid + (1/3)*(conv + bias))
template <int MODE>
__global__ void __launch_bounds__(256)
conv3x3_bf16x3_kernel(const float*  __restrict__ x,     // [B,C,H,W] fp32
                      const __bf16* __restrict__ wph,   // packed hi plane
                      const __bf16* __restrict__ wpl,   // packed lo plane
                      const float*  __restrict__ bias,  // [C]
                      const float*  __restrict__ resid, // [B,C,H,W] (MODE>=1)
                      float*        __restrict__ out)   // [B,C,H,W]
{
    __shared__ __align__(16) __bf16 w_hi_s[WROWS * CIP]; // 27648 B
    __shared__ __align__(16) __bf16 w_lo_s[WROWS * CIP]; // 27648 B
    __shared__ __align__(16) __bf16 x_hi_s[XROWS * CIP]; //  4752 B
    __shared__ __align__(16) __bf16 x_lo_s[XROWS * CIP]; //  4752 B  (~63 KiB)

    const int tid  = threadIdx.x;
    const int wave = tid >> 5;          // 0..7 -> co tile of 16
    const int lane = tid & 31;
    const int half = lane >> 4;
    const int l16  = lane & 15;

    const int blk    = blockIdx.x;      // b*256 + h*2 + wseg
    const int wseg   = blk & 1;
    const int h      = (blk >> 1) & (HW - 1);
    const int b      = blk >> 8;
    const int wstart = wseg * TILE_W;

    const float* xin_b = x + b * C_CH * PLANE;

    v8f acc[4] = {{}, {}, {}, {}};      // N-tiles: w 0..15,16..31,32..47,48..63

    // 12 stages: ci chunk (4 x 32) x kernel row r (3); 3 s-taps per stage
    for (int stage = 0; stage < 12; ++stage) {
        const int r   = stage % 3;
        const int cc  = stage / 3;
        const int ci0 = cc * 32;

        // ---- stage weights: 384 rows x 32 ci into pitch-36 LDS -------------
        {
            const __bf16* srcH = wph + (cc * 3 + r) * (3 * C_CH * 32);
            const __bf16* srcL = wpl + (cc * 3 + r) * (3 * C_CH * 32);
            #pragma unroll
            for (int j = 0; j < 12; ++j) {            // 384*8 / 256
                int k    = tid + j * 256;
                int row  = k >> 3;
                int part = (k & 7) * 4;
#if USE_ASYNC_LDS
                __builtin_amdgcn_global_load_async_to_lds_b64(
                    (gasync_ptr)(srcH + row * 32 + part),
                    (lasync_ptr)(&w_hi_s[row * CIP + part]), 0, 0);
                __builtin_amdgcn_global_load_async_to_lds_b64(
                    (gasync_ptr)(srcL + row * 32 + part),
                    (lasync_ptr)(&w_lo_s[row * CIP + part]), 0, 0);
#else
                *(v4bf*)&w_hi_s[row * CIP + part] = *(const v4bf*)&srcH[row * 32 + part];
                *(v4bf*)&w_lo_s[row * CIP + part] = *(const v4bf*)&srcL[row * 32 + part];
#endif
            }
        }
        // ---- stage input row h+r-1 with halo, split to bf16 hi/lo ----------
        {
            const int hh = h + r - 1;
            const bool rowok = (hh >= 0) && (hh < HW);
            for (int idx = tid; idx < XROWS * 32; idx += 256) {
                int ci = idx / XROWS;
                int wi = idx % XROWS;
                int ww = wstart - 1 + wi;
                float v = 0.0f;
                if (rowok && ww >= 0 && ww < HW)
                    v = xin_b[(ci0 + ci) * PLANE + hh * HW + ww];
                __bf16 hi = (__bf16)v;
                __bf16 lo = (__bf16)(v - (float)hi);
                x_hi_s[wi * CIP + ci] = hi;
                x_lo_s[wi * CIP + ci] = lo;
            }
        }
#if USE_ASYNC_LDS
        asm volatile("s_wait_asynccnt 0" ::: "memory");
#endif
        __syncthreads();

        // ---- compute: 3 s-taps x 4 N-tiles x 3 wmmas (K=32 each) -----------
        #pragma unroll
        for (int s = 0; s < 3; ++s) {
            // A (16co x 32ci): elem e -> K = half*8+e (e<8), 16+half*8+(e-8)
            const int rowA = (s * C_CH + wave * 16 + l16) * CIP;
            v16bf ahi = ld_frag16(&w_hi_s[rowA + half * 8],
                                  &w_hi_s[rowA + 16 + half * 8]);
            v16bf alo = ld_frag16(&w_lo_s[rowA + half * 8],
                                  &w_lo_s[rowA + 16 + half * 8]);
            #pragma unroll
            for (int nt = 0; nt < 4; ++nt) {
                // B (32ci x 16w): elem e -> K = half*8+e (e<8), 16+half*8+(e-8)
                const int rowB = (nt * 16 + l16 + s) * CIP + half * 8;
                v16bf bhi = ld_frag16(&x_hi_s[rowB], &x_hi_s[rowB + 16]);
                v16bf blo = ld_frag16(&x_lo_s[rowB], &x_lo_s[rowB + 16]);
                acc[nt] = __builtin_amdgcn_wmma_f32_16x16x32_bf16(
                    false, ahi, false, bhi, (short)0, acc[nt], false, false);
                acc[nt] = __builtin_amdgcn_wmma_f32_16x16x32_bf16(
                    false, ahi, false, blo, (short)0, acc[nt], false, false);
                acc[nt] = __builtin_amdgcn_wmma_f32_16x16x32_bf16(
                    false, alo, false, bhi, (short)0, acc[nt], false, false);
            }
        }
        __syncthreads();
    }

    // ---- epilogue: D VGPR v -> co = wave*16 + half*8 + v, w = wstart + l16 -
    const float h_step = 1.0f / 3.0f;
    const int outbase = b * C_CH * PLANE + h * HW;
    #pragma unroll
    for (int v = 0; v < 8; ++v) {
        const int co = wave * 16 + half * 8 + v;
        const float bz = bias[co];
        #pragma unroll
        for (int nt = 0; nt < 4; ++nt) {
            const int o0 = outbase + co * PLANE + wstart + nt * 16 + l16;
            float rr = acc[nt][v] + bz;
            if (MODE == 0) {
                out[o0] = fmaxf(rr, 0.0f);
            } else {
                float y = resid[o0] + h_step * rr;
                if (MODE == 2) y = fmaxf(y, 0.0f);
                out[o0] = y;
            }
        }
    }
}

extern "C" void kernel_launch(void* const* d_in, const int* in_sizes, int n_in,
                              void* d_out, int out_size, void* d_ws, size_t ws_size,
                              hipStream_t stream) {
    (void)in_sizes; (void)n_in; (void)out_size; (void)ws_size;
    const float* x  = (const float*)d_in[0];
    const float* w1 = (const float*)d_in[1];
    const float* b1 = (const float*)d_in[2];
    const float* w2 = (const float*)d_in[3];
    const float* b2 = (const float*)d_in[4];
    float* out = (float*)d_out;

    const int NACT = BATCH * C_CH * PLANE;            // 33554432

    float*  tbuf = (float*)d_ws;                      // hidden activation t
    float*  xbuf = tbuf + NACT;                       // x ping-pong
    __bf16* wp1h = (__bf16*)(xbuf + NACT);
    __bf16* wp1l = wp1h + NWEL;
    __bf16* wp2h = wp1l + NWEL;
    __bf16* wp2l = wp2h + NWEL;

    repack_w_bf16_kernel<<<(NWEL + 255) / 256, 256, 0, stream>>>(w1, wp1h, wp1l);
    repack_w_bf16_kernel<<<(NWEL + 255) / 256, 256, 0, stream>>>(w2, wp2h, wp2l);

    dim3 grid(BATCH * HW * (HW / TILE_W));            // 4096 blocks
    dim3 blk(256);

    // step 1
    conv3x3_bf16x3_kernel<0><<<grid, blk, 0, stream>>>(x,    wp1h, wp1l, b1, nullptr, tbuf);
    conv3x3_bf16x3_kernel<1><<<grid, blk, 0, stream>>>(tbuf, wp2h, wp2l, b2, x,       xbuf);
    // step 2
    conv3x3_bf16x3_kernel<0><<<grid, blk, 0, stream>>>(xbuf, wp1h, wp1l, b1, nullptr, tbuf);
    conv3x3_bf16x3_kernel<1><<<grid, blk, 0, stream>>>(tbuf, wp2h, wp2l, b2, xbuf,    out);
    // step 3 (final relu fused; residual read/write is same-index elementwise,
    // so in-place on d_out is race-free)
    conv3x3_bf16x3_kernel<0><<<grid, blk, 0, stream>>>(out,  wp1h, wp1l, b1, nullptr, tbuf);
    conv3x3_bf16x3_kernel<2><<<grid, blk, 0, stream>>>(tbuf, wp2h, wp2l, b2, out,     out);
}
